// AtlasMemoryLayer_80135499809178
// MI455X (gfx1250) — matrix operand order repaired and checked by reading, not compiled, not run on hardware
//
#include <hip/hip_runtime.h>
#include <hip/hip_bf16.h>

typedef _Float16 f16;
typedef __attribute__((ext_vector_type(16))) _Float16 v16h;
typedef __attribute__((ext_vector_type(8)))  float    v8f;

#define B_   2
#define T_   512
#define C_   1024
#define H_   16
#define D_   64
#define CS_  64
#define NC_  8
#define OMW_ 8

// Polar Express quintic coefficients (first NS_STEPS=5)
__constant__ float PE_A[5] = {8.28721201814563f, 4.107059111542203f,
                              3.9486908534822946f, 3.3184196573706015f,
                              2.300652019954817f};
__constant__ float PE_B[5] = {-23.595886519098837f, -2.9478499167379106f,
                              -2.908902115962949f, -2.488488024314874f,
                              -1.6689039845747493f};
__constant__ float PE_C[5] = {17.300387312530933f, 0.5448431082926601f,
                              0.5518191394370137f, 0.51004894012372f,
                              0.4188073119525673f};

// ---------------- WMMA fragment helpers (wave32, 16x16x32 f16) --------------
// A fragment: 16x32 f16, row-major (lda elems). Lane<16: row=lane, K in
// {0..7,16..23}; lane>=16: row=lane-16, K in {8..15,24..31}. 2 f16 per VGPR.
__device__ inline v16h load_frag_a(const f16* A, int lda, int m0, int k0, int lane) {
    int r  = m0 + (lane & 15);
    int kb = k0 + ((lane & 16) ? 8 : 0);
    const f16* p = A + (size_t)r * lda + kb;
    v16h a;
#pragma unroll
    for (int v = 0; v < 8; ++v) {
        int kk = ((v & 4) ? 16 : 0) + 2 * (v & 3);
        a[2 * v]     = p[kk];
        a[2 * v + 1] = p[kk + 1];
    }
    return a;
}

// B fragment from Bt (N x K row-major, i.e. B[k][n] = Bt[n][k]).
// Lane<16: col=lane, K in {0..7,16..23}; lane>=16: col=lane-16, K in {8..15,24..31}.
__device__ inline v16h load_frag_b(const f16* Bt, int ldb, int n0, int k0, int lane) {
    int n  = n0 + (lane & 15);
    int kb = k0 + ((lane & 16) ? 8 : 0);
    const f16* p = Bt + (size_t)n * ldb + kb;
    v16h b;
#pragma unroll
    for (int v = 0; v < 8; ++v) {
        int kk = ((v & 4) ? 16 : 0) + 2 * (v & 3);
        b[2 * v]     = p[kk];
        b[2 * v + 1] = p[kk + 1];
    }
    return b;
}

// D/C layout: lane<16: col=lane, rows r ; lane>=16: col=lane-16, rows 8+r.
__device__ inline void store_acc(float* C, int ldc, int m0, int n0, int lane, v8f acc) {
    int col = n0 + (lane & 15);
    int rb  = m0 + ((lane & 16) ? 8 : 0);
#pragma unroll
    for (int r = 0; r < 8; ++r) C[(size_t)(rb + r) * ldc + col] = acc[r];
}

// Dual-tile 64x64x64 block GEMM: 8 waves, wave computes tiles (m0,n0) and
// (m0+32,n0) sharing B fragments. Accumulators returned in registers so the
// caller can fuse the epilogue (and hold them across a barrier).
__device__ inline void block_gemm2(const f16* A, int lda, const f16* Bt, int ldb,
                                   v8f& acc0, v8f& acc1) {
    int wave = threadIdx.x >> 5;
    int lane = threadIdx.x & 31;
    int m0 = (wave >> 2) << 4;          // 0 or 16
    int n0 = (wave & 3) << 4;           // 0,16,32,48
#pragma unroll
    for (int k0 = 0; k0 < 64; k0 += 32) {
        v16h b  = load_frag_b(Bt, ldb, n0, k0, lane);
        v16h a0 = load_frag_a(A, lda, m0, k0, lane);
        v16h a1 = load_frag_a(A, lda, m0 + 32, k0, lane);
        acc0 = __builtin_amdgcn_wmma_f32_16x16x32_f16(false, a0, false, b,
                                                      (short)0, acc0, false, false);
        acc1 = __builtin_amdgcn_wmma_f32_16x16x32_f16(false, a1, false, b,
                                                      (short)0, acc1, false, false);
    }
}

// ---------------- Kernels ---------------------------------------------------

__global__ void cvt_f32_f16(const float* __restrict__ in, f16* __restrict__ out, int n) {
    int i = blockIdx.x * blockDim.x + threadIdx.x;
    if (i < n) out[i] = (f16)in[i];
}

__global__ void zero_f32(float* __restrict__ p, int n) {
    int i = blockIdx.x * blockDim.x + threadIdx.x;
    if (i < n) p[i] = 0.f;
}

// C[M,N] = A[M,K] * Bt[N,K]^T ; 256 threads per block, block owns a 64x64
// output tile, each wave two 16x16 tiles sharing B fragments.
__global__ void gemm_wmma(const f16* __restrict__ A, const f16* __restrict__ Bt,
                          float* __restrict__ C, int K, int ldc) {
    int wave = threadIdx.x >> 5, lane = threadIdx.x & 31;
    int m0 = blockIdx.y * 64 + ((wave >> 2) << 4);
    int n0 = blockIdx.x * 64 + ((wave & 3) << 4);
    v8f acc0 = {}, acc1 = {};
    for (int k0 = 0; k0 < K; k0 += 32) {
        // prefetch next k-step fragment rows into cache (global_prefetch_b8)
        if (k0 + 32 < K) {
            __builtin_prefetch(A + (size_t)(m0 + (lane & 15)) * K + k0 + 32, 0, 1);
            __builtin_prefetch(A + (size_t)(m0 + 32 + (lane & 15)) * K + k0 + 32, 0, 1);
            __builtin_prefetch(Bt + (size_t)(n0 + (lane & 15)) * K + k0 + 32, 0, 1);
        }
        v16h b  = load_frag_b(Bt, K, n0, k0, lane);
        v16h a0 = load_frag_a(A, K, m0, k0, lane);
        v16h a1 = load_frag_a(A, K, m0 + 32, k0, lane);
        acc0 = __builtin_amdgcn_wmma_f32_16x16x32_f16(false, a0, false, b,
                                                      (short)0, acc0, false, false);
        acc1 = __builtin_amdgcn_wmma_f32_16x16x32_f16(false, a1, false, b,
                                                      (short)0, acc1, false, false);
    }
    store_acc(C, ldc, m0, n0, lane, acc0);
    store_acc(C, ldc, m0 + 32, n0, lane, acc1);
}

// Causal depthwise conv (K=4) + per-head RMS norm + poly features.
// grid (H, T, B), 64 threads (= d). mode 0: norm+poly (q/k), 1: raw (v).
__global__ void conv_norm_poly(const float* __restrict__ pre, const float* __restrict__ w,
                               const float* __restrict__ bias, float* __restrict__ outf,
                               f16* __restrict__ outh, int mode) {
    __shared__ float red[64];
    int d = threadIdx.x, h = blockIdx.x, t = blockIdx.y, b = blockIdx.z;
    int cch = h * D_ + d;
    float val = bias[cch];
#pragma unroll
    for (int j = 0; j < 4; ++j) {
        int ts = t + j - 3;
        float xv = (ts >= 0) ? pre[((size_t)(b * T_ + ts)) * C_ + cch] : 0.f;
        val += w[cch * 4 + j] * xv;
    }
    if (mode == 0) {
        red[d] = val * val;
        __syncthreads();
        for (int s = 32; s > 0; s >>= 1) {
            if (d < s) red[d] += red[d + s];
            __syncthreads();
        }
        float rn = val * rsqrtf(red[0] * (1.f / 64.f) + 1e-6f);
        val = rn + 0.5f * rn * rn;
    }
    size_t o = (((size_t)(b * H_ + h)) * T_ + t) * D_ + d;
    outf[o] = val;
    if (outh) outh[o] = (f16)val;
}

// Gates: sigmoid(x @ W^T) for Wa/We/Wt/Wg. grid B*T, 64 threads.
__global__ void gates_kernel(const float* __restrict__ x,
                             const float* __restrict__ Wa, const float* __restrict__ We,
                             const float* __restrict__ Wt, const float* __restrict__ Wg,
                             float* __restrict__ oa, float* __restrict__ oe,
                             float* __restrict__ ot, float* __restrict__ og) {
    int bt = blockIdx.x;
    int tid = threadIdx.x;
    int which = tid >> 4, h = tid & 15;
    const float* W = (which == 0) ? Wa : (which == 1) ? We : (which == 2) ? Wt : Wg;
    float* O = (which == 0) ? oa : (which == 1) ? oe : (which == 2) ? ot : og;
    const float* xr = x + (size_t)bt * C_;
    const float* wr = W + (size_t)h * C_;
    float acc = 0.f;
    for (int kk = 0; kk < C_; ++kk) acc += xr[kk] * wr[kk];
    O[(size_t)bt * H_ + h] = 1.f / (1.f + __expf(-acc));
}

// Per-(b,h) chunk work: pred = k_c M^T (WMMA), err, omega-window outer
// products, momentum scan -> chunk_S (global). grid B*H, 256 threads.
__global__ void chunk_inner(const f16* __restrict__ kh, const float* __restrict__ kf,
                            const float* __restrict__ vf, const float* __restrict__ ggam,
                            const float* __restrict__ geta, const float* __restrict__ gth,
                            float* __restrict__ Mst, float* __restrict__ Sst,
                            float* __restrict__ chunkS, int c) {
    __shared__ f16  Mh[D_ * D_];
    __shared__ float errs[CS_ * D_];
    __shared__ float kls[CS_ * D_];
    __shared__ float Sl[D_ * D_];
    __shared__ float gmw[CS_], etw[CS_], thw[CS_];
    int bh = blockIdx.x, tid = threadIdx.x;
    int b = bh >> 4, h = bh & 15;
    int wave = tid >> 5, lane = tid & 31;
    int m0 = (wave >> 2) << 4, n0 = (wave & 3) << 4;
    size_t stoff = (size_t)bh * D_ * D_;
    size_t chtd  = (((size_t)bh) * T_ + c * CS_) * D_;
    for (int e = tid; e < D_ * D_; e += 256) {
        Mh[e]  = (f16)Mst[stoff + e];
        Sl[e]  = Sst[stoff + e];
        kls[e] = kf[chtd + e];
    }
    if (tid < CS_) {
        size_t gi = ((size_t)(b * T_ + c * CS_ + tid)) * H_ + h;
        gmw[tid] = ggam[gi];
        etw[tid] = geta[gi];
        thw[tid] = gth[gi];
    }
    __syncthreads();
    // pred[t,i] = sum_k k_c[t,k] * M[i,k]
    {
        v8f p0 = {}, p1 = {};
        block_gemm2(kh + chtd, D_, Mh, D_, p0, p1);
        store_acc(errs, D_, m0, n0, lane, p0);
        store_acc(errs, D_, m0 + 32, n0, lane, p1);
    }
    __syncthreads();
    for (int e = tid; e < CS_ * D_; e += 256) errs[e] -= vf[chtd + e];
    __syncthreads();
    int i = tid >> 2, j0 = (tid & 3) << 4;
    for (int t = 0; t < CS_; ++t) {
        float acc[16];
#pragma unroll
        for (int jj = 0; jj < 16; ++jj) acc[jj] = 0.f;
        int w0 = t - (OMW_ - 1); if (w0 < 0) w0 = 0;
        for (int w = w0; w <= t; ++w) {
            float gei = 2.f * gmw[w] * errs[w * D_ + i];
#pragma unroll
            for (int jj = 0; jj < 16; ++jj) acc[jj] += gei * kls[w * D_ + j0 + jj];
        }
        float th = thw[t], et = etw[t];
        float* srow = Sl + i * D_ + j0;
        float* crow = chunkS + (((size_t)bh * CS_ + t) * D_ * D_) + i * D_ + j0;
#pragma unroll
        for (int jj = 0; jj < 16; ++jj) {
            float s = th * srow[jj] - et * acc[jj];
            srow[jj] = s;
            crow[jj] = s;
        }
    }
    __syncthreads();
    for (int e = tid; e < D_ * D_; e += 256) Sst[stoff + e] = Sl[e];
}

// Polar Express orthogonalization of one 64x64 matrix per block (in place).
// 5 iterations x 3 WMMA 64^3 GEMMs in LDS with fused f16 epilogues.
// LDS = 41 KB. grid B*H*CS, 256 threads.
__global__ void polar_express_kernel(float* __restrict__ chunkS) {
    __shared__ f16   Xh[4096];
    __shared__ f16   XhT[4096];
    __shared__ float Af[4096];
    __shared__ f16   Ah[4096];
    __shared__ float red[256];
    int tid = threadIdx.x;
    int wave = tid >> 5, lane = tid & 31;
    int m0 = (wave >> 2) << 4, n0 = (wave & 3) << 4;
    int col = n0 + (lane & 15);
    int rb0 = m0 + ((lane & 16) ? 8 : 0);
    int rb1 = rb0 + 32;
    size_t base = (size_t)blockIdx.x * 4096;
    float ss = 0.f;
    for (int e = tid; e < 4096; e += 256) {
        float v = chunkS[base + e];
        Af[e] = v;                     // stage in Af temporarily
        ss += v * v;
    }
    red[tid] = ss;
    __syncthreads();
    for (int s = 128; s > 0; s >>= 1) {
        if (tid < s) red[tid] += red[tid + s];
        __syncthreads();
    }
    float scale = 1.f / ((sqrtf(red[0]) + 1e-7f) * 1.01f);
    for (int e = tid; e < 4096; e += 256) {
        float v = Af[e] * scale;
        Xh[e] = (f16)v;
        XhT[((e & 63) << 6) | (e >> 6)] = (f16)v;
    }
    for (int it = 0; it < 5; ++it) {
        float a = PE_A[it], bc = PE_B[it], cc = PE_C[it];
        __syncthreads();                               // X (Xh/XhT) ready
        // A = X X^T ; epilogue writes Af (f32) + Ah (f16) directly
        {
            v8f a0 = {}, a1 = {};
            block_gemm2(Xh, 64, Xh, 64, a0, a1);
#pragma unroll
            for (int r = 0; r < 8; ++r) {
                int i0 = (rb0 + r) * 64 + col;
                int i1 = (rb1 + r) * 64 + col;
                Af[i0] = a0[r]; Ah[i0] = (f16)a0[r];
                Af[i1] = a1[r]; Ah[i1] = (f16)a1[r];
            }
        }
        __syncthreads();                               // A ready
        // A2 = A·A (A symmetric) ; keep accs in regs across barrier, then
        // overwrite Ah in place with P = b*A + c*A2 (f16)
        {
            v8f p0 = {}, p1 = {};
            block_gemm2(Ah, 64, Ah, 64, p0, p1);
            __syncthreads();                           // all Ah reads done
#pragma unroll
            for (int r = 0; r < 8; ++r) {
                int i0 = (rb0 + r) * 64 + col;
                int i1 = (rb1 + r) * 64 + col;
                Ah[i0] = (f16)(bc * Af[i0] + cc * p0[r]);
                Ah[i1] = (f16)(bc * Af[i1] + cc * p1[r]);
            }
        }
        __syncthreads();                               // P ready
        // X = a*X + P @ X ; accs in regs across barrier, then update Xh/XhT
        {
            v8f x0 = {}, x1 = {};
            block_gemm2(Ah, 64, XhT, 64, x0, x1);
            __syncthreads();                           // all Xh/XhT reads done
#pragma unroll
            for (int r = 0; r < 8; ++r) {
                int i0 = (rb0 + r) * 64 + col;
                float xn0 = a * (float)Xh[i0] + x0[r];
                Xh[i0] = (f16)xn0;
                XhT[col * 64 + rb0 + r] = (f16)xn0;
                int i1 = (rb1 + r) * 64 + col;
                float xn1 = a * (float)Xh[i1] + x1[r];
                Xh[i1] = (f16)xn1;
                XhT[col * 64 + rb1 + r] = (f16)xn1;
            }
        }
    }
    __syncthreads();
    for (int e = tid; e < 4096; e += 256) chunkS[base + e] = (float)Xh[e];
}

// Memory scan: M = alpha_t*M + X_orth_t ; y_t = M q_t. grid B*H, 256 threads.
__global__ void memory_scan(const float* __restrict__ qf, const float* __restrict__ gal,
                            const float* __restrict__ chunkS, float* __restrict__ Mst,
                            float* __restrict__ yf, int c) {
    __shared__ float Ml[4096];
    __shared__ float qs[CS_ * D_];
    __shared__ float alw[CS_];
    int bh = blockIdx.x, tid = threadIdx.x;
    int b = bh >> 4, h = bh & 15;
    size_t stoff = (size_t)bh * 4096;
    size_t chtd  = (((size_t)bh) * T_ + c * CS_) * D_;
    for (int e = tid; e < 4096; e += 256) {
        Ml[e] = Mst[stoff + e];
        qs[e] = qf[chtd + e];
    }
    if (tid < CS_) alw[tid] = gal[((size_t)(b * T_ + c * CS_ + tid)) * H_ + h];
    __syncthreads();
    for (int t = 0; t < CS_; ++t) {
        float al = alw[t];
        const float* xo = chunkS + ((size_t)bh * CS_ + t) * 4096;
        for (int e = tid; e < 4096; e += 256) Ml[e] = al * Ml[e] + xo[e];
        __syncthreads();
        if (tid < D_) {
            float acc = 0.f;
            const float* mr = Ml + tid * D_;
            const float* qr = qs + t * D_;
            for (int kk = 0; kk < D_; ++kk) acc += mr[kk] * qr[kk];
            yf[chtd + (size_t)t * D_ + tid] = acc;
        }
        __syncthreads();
    }
    for (int e = tid; e < 4096; e += 256) Mst[stoff + e] = Ml[e];
}

// Final per-head RMS norm of y, reshaped to (B*T, C) in f16.
__global__ void out_norm(const float* __restrict__ yf, f16* __restrict__ yh) {
    __shared__ float red[64];
    int d = threadIdx.x, h = blockIdx.x, t = blockIdx.y, b = blockIdx.z;
    size_t o = (((size_t)(b * H_ + h)) * T_ + t) * D_ + d;
    float v = yf[o];
    red[d] = v * v;
    __syncthreads();
    for (int s = 32; s > 0; s >>= 1) {
        if (d < s) red[d] += red[d + s];
        __syncthreads();
    }
    float rn = v * rsqrtf(red[0] * (1.f / 64.f) + 1e-6f);
    yh[((size_t)(b * T_ + t)) * C_ + h * D_ + d] = (f16)rn;
}

// ---------------- Launch ----------------------------------------------------
extern "C" void kernel_launch(void* const* d_in, const int* in_sizes, int n_in,
                              void* d_out, int out_size, void* d_ws, size_t ws_size,
                              hipStream_t stream) {
    (void)in_sizes; (void)n_in; (void)out_size; (void)ws_size;
    const float* x   = (const float*)d_in[0];
    const float* Wq  = (const float*)d_in[1];
    const float* Wk  = (const float*)d_in[2];
    const float* Wv  = (const float*)d_in[3];
    const float* Wo  = (const float*)d_in[4];
    const float* cqw = (const float*)d_in[5];
    const float* cqb = (const float*)d_in[6];
    const float* ckw = (const float*)d_in[7];
    const float* ckb = (const float*)d_in[8];
    const float* cvw = (const float*)d_in[9];
    const float* cvb = (const float*)d_in[10];
    const float* Wa  = (const float*)d_in[11];
    const float* We  = (const float*)d_in[12];
    const float* Wt  = (const float*)d_in[13];
    const float* Wg  = (const float*)d_in[14];
    float* out = (float*)d_out;

    char* ws = (char*)d_ws;
    size_t off = 0;
    auto take = [&](size_t bytes) -> char* {
        char* p = ws + off;
        off = (off + bytes + 255) & ~(size_t)255;
        return p;
    };
    const size_t MT = (size_t)B_ * T_;               // 1024 rows
    f16*   xh   = (f16*)take(MT * C_ * 2);
    f16*   wqh  = (f16*)take((size_t)C_ * C_ * 2);
    f16*   wkh  = (f16*)take((size_t)C_ * C_ * 2);
    f16*   wvh  = (f16*)take((size_t)C_ * C_ * 2);
    f16*   woh  = (f16*)take((size_t)C_ * C_ * 2);
    float* preq = (float*)take(MT * C_ * 4);
    float* prek = (float*)take(MT * C_ * 4);
    float* prev = (float*)take(MT * C_ * 4);
    float* qfp  = (float*)take(MT * C_ * 4);         // (B,H,T,D) f32
    f16*   khp  = (f16*)take(MT * C_ * 2);           // (B,H,T,D) f16
    float* kfp  = (float*)take(MT * C_ * 4);
    float* vfp  = (float*)take(MT * C_ * 4);
    float* gA   = (float*)take(MT * H_ * 4);
    float* gE   = (float*)take(MT * H_ * 4);
    float* gT   = (float*)take(MT * H_ * 4);
    float* gG   = (float*)take(MT * H_ * 4);
    float* Mst  = (float*)take((size_t)B_ * H_ * D_ * D_ * 4);
    float* Sst  = (float*)take((size_t)B_ * H_ * D_ * D_ * 4);
    float* chS  = (float*)take((size_t)B_ * H_ * CS_ * D_ * D_ * 4);  // 32 MB
    float* yfp  = (float*)take(MT * C_ * 4);
    f16*   yh   = (f16*)take(MT * C_ * 2);

    // 1) precision converts
    int nx = (int)(MT * C_);
    cvt_f32_f16<<<dim3((nx + 255) / 256), dim3(256), 0, stream>>>(x, xh, nx);
    int nw = C_ * C_;
    cvt_f32_f16<<<dim3((nw + 255) / 256), dim3(256), 0, stream>>>(Wq, wqh, nw);
    cvt_f32_f16<<<dim3((nw + 255) / 256), dim3(256), 0, stream>>>(Wk, wkh, nw);
    cvt_f32_f16<<<dim3((nw + 255) / 256), dim3(256), 0, stream>>>(Wv, wvh, nw);
    cvt_f32_f16<<<dim3((nw + 255) / 256), dim3(256), 0, stream>>>(Wo, woh, nw);

    // 2) projection GEMMs (WMMA, 64x64 block tiles)
    dim3 gemm_grid(C_ / 64, (int)MT / 64);
    gemm_wmma<<<gemm_grid, 256, 0, stream>>>(xh, wqh, preq, C_, C_);
    gemm_wmma<<<gemm_grid, 256, 0, stream>>>(xh, wkh, prek, C_, C_);
    gemm_wmma<<<gemm_grid, 256, 0, stream>>>(xh, wvh, prev, C_, C_);

    // 3) conv + norm + poly
    dim3 cg(H_, T_, B_);
    conv_norm_poly<<<cg, 64, 0, stream>>>(preq, cqw, cqb, qfp, (f16*)nullptr, 0);
    conv_norm_poly<<<cg, 64, 0, stream>>>(prek, ckw, ckb, kfp, khp, 0);
    conv_norm_poly<<<cg, 64, 0, stream>>>(prev, cvw, cvb, vfp, (f16*)nullptr, 1);

    // 4) gates
    gates_kernel<<<dim3((int)MT), 64, 0, stream>>>(x, Wa, We, Wt, Wg, gA, gE, gT, gG);

    // 5) init recurrent state
    int nst = B_ * H_ * D_ * D_;
    zero_f32<<<dim3((nst + 255) / 256), dim3(256), 0, stream>>>(Mst, nst);
    zero_f32<<<dim3((nst + 255) / 256), dim3(256), 0, stream>>>(Sst, nst);

    // 6) sequential chunk loop
    for (int c = 0; c < NC_; ++c) {
        chunk_inner<<<dim3(B_ * H_), 256, 0, stream>>>(khp, kfp, vfp, gG, gE, gT,
                                                       Mst, Sst, chS, c);
        polar_express_kernel<<<dim3(B_ * H_ * CS_), 256, 0, stream>>>(chS);
        memory_scan<<<dim3(B_ * H_), 256, 0, stream>>>(qfp, gA, chS, Mst, yfp, c);
    }

    // 7) output norm + projection
    out_norm<<<cg, 64, 0, stream>>>(yfp, yh);
    gemm_wmma<<<gemm_grid, 256, 0, stream>>>(yh, woh, out, C_, C_);
}
